// MutilLocalLoss_40561671144147
// MI455X (gfx1250) — matrix operand order repaired and checked by reading, not compiled, not run on hardware
//
#include <hip/hip_runtime.h>
#include <math.h>

// ---------------------------------------------------------------------------
// MultiLevelLocalLoss for MI455X (gfx1250, wave32, WMMA).
// Correlation recast as GEMM:  R[r,i] = sum_{c,j} sat[b,c,r,x+j]*g[b,c,i,j]
//   M = A (sat rows), N = cropH (template rows), K = C*cropW
//   corr[y,x] = sum_i R[y+i, i]   (diagonal band sum in LDS)
// K-blocked LDS-staged pipeline feeding v_wmma_f32_16x16x32_bf16.
// ---------------------------------------------------------------------------

typedef __attribute__((ext_vector_type(16))) __bf16 v16bf;
typedef __attribute__((ext_vector_type(8)))  __bf16 v8bf;
typedef __attribute__((ext_vector_type(8)))  float  v8f;

#define NB 8  // batch size

// --------------------------- sat repack to bf16 ----------------------------
// input layout (B,C,A,A) f32  ->  Sbf layout [b][r][c][col] bf16
template<int A, int C>
__global__ __launch_bounds__(256)
void sat_convert_kernel(const float* __restrict__ sat, __bf16* __restrict__ Sbf, int total)
{
    for (int idx = blockIdx.x * blockDim.x + threadIdx.x; idx < total;
         idx += gridDim.x * blockDim.x) {
        int col = idx % A;
        int t1  = idx / A;
        int r   = t1 % A;
        int t2  = t1 / A;
        int c   = t2 % C;
        int b   = t2 / C;
        Sbf[(((size_t)b * A + r) * C + c) * A + col] = (__bf16)sat[idx];
    }
}

// ------------------- template: mask*crop, normalize, pad -------------------
// Gbf layout [b][NPAD][KPAD] bf16, k = c*CROPW + j, zero padded.
template<int A, int C, int CROPH, int CROPW, int TOP, int KPAD, int NPAD>
__global__ __launch_bounds__(256)
void prep_template_kernel(const float* __restrict__ grd, const float* __restrict__ mask,
                          __bf16* __restrict__ Gbf)
{
    constexpr int K = C * CROPW;
    const int b = blockIdx.x;
    __shared__ float red[256];
    __shared__ float s_inv;

    float ss = 0.f;
    for (int idx = threadIdx.x; idx < C * CROPH * CROPW; idx += blockDim.x) {
        int c = idx / (CROPH * CROPW);
        int r = idx % (CROPH * CROPW);
        int i = r / CROPW, j = r % CROPW;
        float mv = mask[(TOP + i) * A + (TOP + j)];
        float v  = grd[(((size_t)b * C + c) * A + (TOP + i)) * A + (TOP + j)] * mv;
        ss += v * v;
    }
    red[threadIdx.x] = ss;
    __syncthreads();
    for (int s = 128; s > 0; s >>= 1) {
        if ((int)threadIdx.x < s) red[threadIdx.x] += red[threadIdx.x + s];
        __syncthreads();
    }
    if (threadIdx.x == 0) {
        float norm = sqrtf(red[0]);
        s_inv = 1.0f / fmaxf(norm, 1e-12f);
    }
    __syncthreads();
    const float inv = s_inv;

    for (int idx = threadIdx.x; idx < NPAD * KPAD; idx += blockDim.x) {
        int i = idx / KPAD;
        int k = idx % KPAD;
        float v = 0.f;
        if (i < CROPH && k < K) {
            int c = k / CROPW, j = k % CROPW;
            float mv = mask[(TOP + i) * A + (TOP + j)];
            v = grd[(((size_t)b * C + c) * A + (TOP + i)) * A + (TOP + j)] * mv * inv;
        }
        Gbf[((size_t)b * NPAD + i) * KPAD + k] = (__bf16)v;
    }
}

// ----------------------- denominator: sqrt(sum sat^2 * mask) ---------------
template<int A, int C>
__global__ __launch_bounds__(256)
void denom_kernel(const float* __restrict__ sat, const float* __restrict__ mask,
                  float* __restrict__ denom)
{
    const int b = blockIdx.x;
    __shared__ float red[256];
    float ss = 0.f;
    for (int idx = threadIdx.x; idx < C * A * A; idx += blockDim.x) {
        int hw = idx % (A * A);
        float v = sat[(size_t)b * C * A * A + idx];
        ss += v * v * mask[hw];
    }
    red[threadIdx.x] = ss;
    __syncthreads();
    for (int s = 128; s > 0; s >>= 1) {
        if ((int)threadIdx.x < s) red[threadIdx.x] += red[threadIdx.x + s];
        __syncthreads();
    }
    if (threadIdx.x == 0) denom[b] = fmaxf(sqrtf(red[0] + 1e-8f), 1e-6f);
}

// --------------------------- WMMA correlation ------------------------------
// One workgroup (4 waves) per (b, x). K-blocked: stage A/B 32-wide K panels
// in LDS, all waves run their (unrolled) tile set with live accumulators,
// then band-sum corr[y,x] = sum_i R[y+i][i] and apply 2 - 2*corr/denom.
template<int A, int C, int CROPH, int CROPW, int OUTH, int OUTW, int KPAD, int NPAD>
__global__ __launch_bounds__(128)
void corr_wmma_kernel(const __bf16* __restrict__ Sbf, const __bf16* __restrict__ Gbf,
                      const float* __restrict__ denom, float* __restrict__ corr)
{
    constexpr int K      = C * CROPW;
    constexpr int MT     = A / 16;
    constexpr int NT     = NPAD / 16;
    constexpr int NTILES = MT * NT;
    constexpr int NWAVES = 4;
    constexpr int TPW    = (NTILES + NWAVES - 1) / NWAVES;  // tiles per wave
    constexpr int AITER  = (A * 32) / 128;                  // A-panel stage iters

    __shared__ __attribute__((aligned(32))) __bf16 LA[A][32];     // im2col A panel
    __shared__ __attribute__((aligned(32))) __bf16 LB[NPAD][32];  // template B panel
    __shared__ float R[A][NPAD];

    const int b    = blockIdx.x / OUTW;
    const int x    = blockIdx.x % OUTW;
    const int wave = threadIdx.x >> 5;
    const int lane = threadIdx.x & 31;
    const int hi   = lane >> 4;     // 0: lanes 0-15, 1: lanes 16-31
    const int l15  = lane & 15;

    const __bf16* __restrict__ Sb = Sbf + (size_t)b * A * C * A;
    const __bf16* __restrict__ Gb = Gbf + (size_t)b * NPAD * KPAD;

    v8f acc[TPW];
    #pragma unroll
    for (int u = 0; u < TPW; ++u) acc[u] = (v8f){};

    for (int kk = 0; kk < KPAD; kk += 32) {
        // ---- stage im2col A panel: LA[r][kl] = sat[b, c(kg), r, x + j(kg)] ----
        #pragma unroll
        for (int it = 0; it < AITER; ++it) {
            int idx = threadIdx.x + it * 128;
            int r  = idx >> 5;
            int kl = idx & 31;
            int kg = kk + kl;
            __bf16 v = (__bf16)0.f;
            if (kg < K) {                       // only trims the last padded block
                int c = kg / CROPW;
                int j = kg - c * CROPW;
                v = Sb[((size_t)r * C + c) * A + (x + j)];
            }
            LA[r][kl] = v;
        }
        // ---- stage B panel with 8-byte copies (Gb rows are K-contiguous) ----
        for (int idx = threadIdx.x; idx < NPAD * 8; idx += 128) {
            int n  = idx >> 3;
            int kq = (idx & 7) * 4;
            *(unsigned long long*)&LB[n][kq] =
                *(const unsigned long long*)&Gb[(size_t)n * KPAD + kk + kq];
        }
        if (kk + 32 < KPAD && (int)threadIdx.x < NPAD)   // warm next template block
            __builtin_prefetch(&Gb[(size_t)threadIdx.x * KPAD + kk + 32], 0, 1);
        __syncthreads();

        // ---- all waves: unrolled tile sweep, fragments via wide LDS loads ----
        #pragma unroll
        for (int u = 0; u < TPW; ++u) {
            int t = wave + u * NWAVES;
            if (t < NTILES) {
                int mt = t / NT, nt = t % NT;
                int m = mt * 16 + l15;          // A-matrix row for this lane
                int n = nt * 16 + l15;          // B-matrix column for this lane
                // A 16x32 bf16 layout: lane half hi reads K hi*16..+15 (contiguous)
                v8bf a0 = *(const v8bf*)&LA[m][hi * 16];
                v8bf a1 = *(const v8bf*)&LA[m][hi * 16 + 8];
                v16bf af = __builtin_shufflevector(a0, a1,
                    0, 1, 2, 3, 4, 5, 6, 7, 8, 9, 10, 11, 12, 13, 14, 15);
                // B 32x16 bf16 layout: lane half hi reads K hi*16..+15 (contiguous)
                v16bf bf = *(const v16bf*)&LB[n][hi * 16];
                acc[u] = __builtin_amdgcn_wmma_f32_16x16x32_bf16(
                    false, af, false, bf, (short)0, acc[u], false, false);
            }
        }
        __syncthreads();
    }

    // ---- spill accumulators to R (C/D layout: VGPR g -> M = g + hi*8) ----
    #pragma unroll
    for (int u = 0; u < TPW; ++u) {
        int t = wave + u * NWAVES;
        if (t < NTILES) {
            int mt = t / NT, nt = t % NT;
            #pragma unroll
            for (int g = 0; g < 8; ++g)
                R[mt * 16 + g + hi * 8][nt * 16 + l15] = acc[u][g];
        }
    }
    __syncthreads();

    const float dinv = 2.0f / denom[b];
    for (int y = threadIdx.x; y < OUTH; y += blockDim.x) {
        float s = 0.f;
        #pragma unroll 4
        for (int i = 0; i < CROPH; ++i) s += R[y + i][i];
        corr[((size_t)b * OUTH + y) * OUTW + x] = 2.0f - dinv * s;
    }
}

// ------------------------ GT position gather -------------------------------
template<int OUTH, int OUTW>
__global__ void pos_kernel(const float* __restrict__ gu, const float* __restrict__ gv,
                           const float* __restrict__ gh, const float* __restrict__ corr,
                           float* __restrict__ pos, float mpp)
{
    int b = threadIdx.x;
    if (b >= NB) return;
    float ang = gh[b] * (float)(10.0 * M_PI / 180.0);
    float cs = cosf(ang), sn = sinf(ang);
    float gdx = -gu[b] * 20.0f;
    float gdy = -gv[b] * 20.0f;
    float dxr = -gdx * cs + gdy * sn;
    float dyr =  gdx * sn + gdy * cs;
    int w = (int)rintf(OUTW * 0.5f - 0.5f + dxr / mpp);   // RNE matches jnp.round
    int h = (int)rintf(OUTH * 0.5f - 0.5f + dyr / mpp);
    w = min(max(w, 0), OUTW - 1);
    h = min(max(h, 0), OUTH - 1);
    pos[b] = corr[((size_t)b * OUTH + h) * OUTW + w];
}

// ---------------- softplus loss partials (deterministic, no atomics) -------
__global__ __launch_bounds__(256)
void loss_partial_kernel(const float* __restrict__ corr, const float* __restrict__ pos,
                         float* __restrict__ partial, int outHW, float scale)
{
    __shared__ float red[256];
    float acc = 0.f;
    const int total = NB * outHW;
    for (int idx = blockIdx.x * blockDim.x + threadIdx.x; idx < total;
         idx += gridDim.x * blockDim.x) {
        int b = idx / outHW;
        float z = (pos[b] - corr[idx]) * 10.0f;
        acc += fmaxf(z, 0.f) + log1pf(expf(-fabsf(z)));   // stable logaddexp(0,z)
    }
    red[threadIdx.x] = acc;
    __syncthreads();
    for (int s = 128; s > 0; s >>= 1) {
        if ((int)threadIdx.x < s) red[threadIdx.x] += red[threadIdx.x + s];
        __syncthreads();
    }
    if (threadIdx.x == 0) partial[blockIdx.x] = red[0] * scale;
}

__global__ void finalize_kernel(const float* __restrict__ partial, float* __restrict__ out)
{
    if (threadIdx.x == 0 && blockIdx.x == 0) {
        float s = 0.f;
        for (int i = 0; i < 4 * 64; ++i) s += partial[i];
        out[0] = s;
    }
}

// ------------------------------- host side ---------------------------------
template<int A, int C, int CROPH, int CROPW, int TOP, int OUTH, int OUTW, int KPAD, int NPAD>
static void run_level(const float* grd, const float* sat, const float* mask,
                      __bf16* Sbf, __bf16* Gbf, float* corr, float* denom, float* pos,
                      const float* gu, const float* gv, const float* gh,
                      float* partial, float mpp, hipStream_t stream)
{
    int totalS = NB * C * A * A;
    sat_convert_kernel<A, C><<<(totalS + 255) / 256, 256, 0, stream>>>(sat, Sbf, totalS);
    prep_template_kernel<A, C, CROPH, CROPW, TOP, KPAD, NPAD><<<NB, 256, 0, stream>>>(grd, mask, Gbf);
    denom_kernel<A, C><<<NB, 256, 0, stream>>>(sat, mask, denom);
    corr_wmma_kernel<A, C, CROPH, CROPW, OUTH, OUTW, KPAD, NPAD>
        <<<NB * OUTW, 128, 0, stream>>>(Sbf, Gbf, denom, corr);
    pos_kernel<OUTH, OUTW><<<1, 32, 0, stream>>>(gu, gv, gh, corr, pos, mpp);
    loss_partial_kernel<<<64, 256, 0, stream>>>(corr, pos, partial, OUTH * OUTW,
                                                1.0f / (NB * (float)(OUTH * OUTW - 1)));
}

extern "C" void kernel_launch(void* const* d_in, const int* in_sizes, int n_in,
                              void* d_out, int out_size, void* d_ws, size_t ws_size,
                              hipStream_t stream)
{
    const float* grd[4]  = {(const float*)d_in[0], (const float*)d_in[1],
                            (const float*)d_in[2], (const float*)d_in[3]};
    const float* sat[4]  = {(const float*)d_in[4], (const float*)d_in[5],
                            (const float*)d_in[6], (const float*)d_in[7]};
    const float* mask[4] = {(const float*)d_in[8], (const float*)d_in[9],
                            (const float*)d_in[10], (const float*)d_in[11]};
    const float* gu = (const float*)d_in[12];
    const float* gv = (const float*)d_in[13];
    const float* gh = (const float*)d_in[14];
    float* out = (float*)d_out;

    // Workspace layout (levels run sequentially on `stream`, buffers reused):
    //   Sbf  : max 8*128*8*128 bf16 = 2 MiB
    //   Gbf  : max 8*64*448  bf16   = 448 KiB  (reserve 512 KiB)
    //   corr : max 8*76*76 f32      = ~181 KiB (reserve 192 KiB)
    //   denom/pos : 8 f32 each;  partials : 4*64 f32
    char* ws = (char*)d_ws;
    __bf16* Sbf   = (__bf16*)(ws + 0);
    __bf16* Gbf   = (__bf16*)(ws + 2u * 1024 * 1024);
    float*  corr  = (float*)(ws + 2u * 1024 * 1024 + 512u * 1024);
    float*  denom = (float*)(ws + 2u * 1024 * 1024 + 512u * 1024 + 192u * 1024);
    float*  pos   = (float*)((char*)denom + 256);
    float*  part  = (float*)((char*)pos + 256);

    //            A    C  cropH cropW top outH outW KPAD NPAD
    run_level< 16, 64,  6,  6,  5, 11, 11, 384, 16>(grd[0], sat[0], mask[0], Sbf, Gbf,
        corr, denom, pos, gu, gv, gh, part + 0 * 64, 6.4f, stream);
    run_level< 32, 32, 13, 13, 10, 20, 20, 416, 16>(grd[1], sat[1], mask[1], Sbf, Gbf,
        corr, denom, pos, gu, gv, gh, part + 1 * 64, 3.2f, stream);
    run_level< 64, 16, 26, 26, 19, 39, 39, 416, 32>(grd[2], sat[2], mask[2], Sbf, Gbf,
        corr, denom, pos, gu, gv, gh, part + 2 * 64, 1.6f, stream);
    run_level<128,  8, 53, 53, 38, 76, 76, 448, 64>(grd[3], sat[3], mask[3], Sbf, Gbf,
        corr, denom, pos, gu, gv, gh, part + 3 * 64, 0.8f, stream);

    finalize_kernel<<<1, 32, 0, stream>>>(part, out);
}
